// MultiheadSelfAttentionEnc_48996986913025
// MI455X (gfx1250) — matrix operand order repaired and checked by reading, not compiled, not run on hardware
//
#include <hip/hip_runtime.h>

// Problem constants: B=1, S=2048, C=2048, H=16, HD=128
#define S_LEN 2048
#define C_DIM 2048
#define HD_DIM 128
#define N_HEAD 16

typedef __attribute__((ext_vector_type(16))) __bf16 bf16x16;
typedef __attribute__((ext_vector_type(8)))  __bf16 bf16x8;
typedef __attribute__((ext_vector_type(8)))  float  f32x8;

static __device__ __forceinline__ f32x8 wmma_bf16(bf16x16 a, bf16x16 b, f32x8 c) {
    // D = A(16x32 bf16) x B(32x16 bf16) + C(16x16 f32)
    return __builtin_amdgcn_wmma_f32_16x16x32_bf16(false, a, false, b, (short)0, c, false, false);
}

// A-fragment (16x32 bf16, rows striped per ISA table):
// lane m (m=lane&15, hi=lane>>4): elems 0..7 = K[k0+hi*8 .. +8), elems 8..15 = K[k0+16+hi*8 .. +8)
static __device__ __forceinline__ bf16x16 load_a_frag(const __bf16* base, int ld, int k0, int lane) {
    int m = lane & 15, hi = lane >> 4;
    const __bf16* p = base + (size_t)m * ld + k0 + hi * 8;
    bf16x16 a;
    bf16x8* h = reinterpret_cast<bf16x8*>(&a);
    h[0] = *reinterpret_cast<const bf16x8*>(p);
    h[1] = *reinterpret_cast<const bf16x8*>(p + 16);
    return a;
}

// B-fragment (32x16 bf16): lane n holds column n, elems 0..7 = K[k0+hi*8 .. +8), elems 8..15 = K[k0+16+hi*8 .. +8)
static __device__ __forceinline__ bf16x16 load_b_frag(const __bf16* base, int ld, int k0, int lane) {
    int n = lane & 15, hi = lane >> 4;
    const __bf16* p = base + (size_t)n * ld + k0 + hi * 8;
    bf16x16 b;
    bf16x8* h = reinterpret_cast<bf16x8*>(&b);
    h[0] = *reinterpret_cast<const bf16x8*>(p);
    h[1] = *reinterpret_cast<const bf16x8*>(p + 16);
    return b;
}

// ---------------------------------------------------------------- fp32 -> bf16
__global__ __launch_bounds__(256) void cvt_f32_bf16(const float* __restrict__ src,
                                                    __bf16* __restrict__ dst, int n8) {
    int i = blockIdx.x * 256 + threadIdx.x;
    if (i >= n8) return;
    const float4* s = reinterpret_cast<const float4*>(src) + (size_t)i * 2;
    float4 v0 = s[0], v1 = s[1];
    bf16x8 o;
    o[0] = (__bf16)v0.x; o[1] = (__bf16)v0.y; o[2] = (__bf16)v0.z; o[3] = (__bf16)v0.w;
    o[4] = (__bf16)v1.x; o[5] = (__bf16)v1.y; o[6] = (__bf16)v1.z; o[7] = (__bf16)v1.w;
    reinterpret_cast<bf16x8*>(dst)[i] = o;
}

// ------------------------------------------------- QKV GEMM + bias + RMSNorm + RoPE
// grid: (S/128, H, 3)  z: 0=Q 1=K 2=V.   block: 256 (8 waves of 32)
// Q/K out: [h][s][d] bf16.  V out: [h][d][s] bf16 (pre-transposed for PV B-fragments).
__global__ __launch_bounds__(256) void qkv_kernel(
    const __bf16* __restrict__ xb,
    const __bf16* __restrict__ wqb, const __bf16* __restrict__ wkb, const __bf16* __restrict__ wvb,
    const float* __restrict__ bq, const float* __restrict__ bk, const float* __restrict__ bv,
    const float* __restrict__ qw, const float* __restrict__ kw,
    const float* __restrict__ rope,
    __bf16* __restrict__ Qb, __bf16* __restrict__ Kb, __bf16* __restrict__ Vt)
{
    __shared__ float tile[128 * 128];

    const int lane = threadIdx.x & 31;
    const int wave = threadIdx.x >> 5;
    const int mt = blockIdx.x, h = blockIdx.y, z = blockIdx.z;

    const __bf16* W    = (z == 0) ? wqb : (z == 1) ? wkb : wvb;
    const float*  bias = (z == 0) ? bq  : (z == 1) ? bk  : bv;

    const __bf16* Arow = xb + (size_t)(mt * 128 + wave * 16) * C_DIM;
    const __bf16* Brow = W  + (size_t)(h * HD_DIM) * C_DIM;

    // Software-pipelined GEMM: double-buffered fragments, k stepped by 64 (2 x 32).
    f32x8 acc[8] = {};
    bf16x16 aF0, bF0[8];
    aF0 = load_a_frag(Arow, C_DIM, 0, lane);
#pragma unroll
    for (int nt = 0; nt < 8; nt++) bF0[nt] = load_b_frag(Brow + (size_t)nt * 16 * C_DIM, C_DIM, 0, lane);

    for (int k0 = 0; k0 < C_DIM; k0 += 64) {
        bf16x16 aF1 = load_a_frag(Arow, C_DIM, k0 + 32, lane);
        bf16x16 bF1[8];
#pragma unroll
        for (int nt = 0; nt < 8; nt++)
            bF1[nt] = load_b_frag(Brow + (size_t)nt * 16 * C_DIM, C_DIM, k0 + 32, lane);
#pragma unroll
        for (int nt = 0; nt < 8; nt++) acc[nt] = wmma_bf16(aF0, bF0[nt], acc[nt]);
        if (k0 + 64 < C_DIM) {
            aF0 = load_a_frag(Arow, C_DIM, k0 + 64, lane);
#pragma unroll
            for (int nt = 0; nt < 8; nt++)
                bF0[nt] = load_b_frag(Brow + (size_t)nt * 16 * C_DIM, C_DIM, k0 + 64, lane);
        }
#pragma unroll
        for (int nt = 0; nt < 8; nt++) acc[nt] = wmma_bf16(aF1, bF1[nt], acc[nt]);
    }

    const int nlane = lane & 15, hi = lane >> 4;

    if (z == 2) {
        // V: columns of the C fragment are single d values; a lane's 8 accs for one
        // n-tile are 8 consecutive s rows -> contiguous bf16x8 store into Vt[h][d][s].
        const float* bvp = bias + h * HD_DIM;
        const int sbase = mt * 128 + wave * 16 + 8 * hi;
        __bf16* dstV = Vt + (size_t)h * HD_DIM * S_LEN;
#pragma unroll
        for (int nt = 0; nt < 8; nt++) {
            int d = nt * 16 + nlane;
            float bb = bvp[d];
            alignas(16) __bf16 ob[8];
#pragma unroll
            for (int v = 0; v < 8; v++) ob[v] = (__bf16)(acc[nt][v] + bb);
            *reinterpret_cast<bf16x8*>(dstV + (size_t)d * S_LEN + sbase) =
                *reinterpret_cast<bf16x8*>(ob);
        }
        return;
    }

    // Q/K: dump C fragments to LDS, then row-wise RMSNorm + RoPE.
#pragma unroll
    for (int nt = 0; nt < 8; nt++)
#pragma unroll
        for (int v = 0; v < 8; v++)
            tile[(wave * 16 + v + 8 * hi) * 128 + nt * 16 + nlane] = acc[nt][v];
    __syncthreads();

    const int r = threadIdx.x >> 1, half = threadIdx.x & 1;
    const int srow = mt * 128 + r;
    const float* rowp = &tile[r * 128 + half * 64];
    const float* bp = bias + h * HD_DIM + half * 64;

    float vals[64];
    float ssq = 0.0f;
#pragma unroll
    for (int j = 0; j < 64; j++) {
        float vv = rowp[j] + bp[j];
        vals[j] = vv;
        ssq += vv * vv;
    }
    ssq += __shfl_xor(ssq, 1, 32);          // partner thread handles other half of row
    float rinv = rsqrtf(ssq * (1.0f / 128.0f) + 1.1920929e-07f);
    const float* nw = ((z == 0) ? qw : kw) + half * 64;
    const float* rp = rope + ((size_t)srow * 64 + half * 32) * 4;  // [s][pair][2][2]
    alignas(16) __bf16 obuf[64];
#pragma unroll
    for (int p = 0; p < 32; p++) {
        float x0 = vals[2 * p]     * rinv * nw[2 * p];
        float x1 = vals[2 * p + 1] * rinv * nw[2 * p + 1];
        float4 rc = *reinterpret_cast<const float4*>(rp + p * 4);
        obuf[2 * p]     = (__bf16)(rc.x * x0 + rc.y * x1);
        obuf[2 * p + 1] = (__bf16)(rc.z * x0 + rc.w * x1);
    }
    __bf16* dst = ((z == 0) ? Qb : Kb) + ((size_t)h * S_LEN + srow) * HD_DIM + half * 64;
#pragma unroll
    for (int c = 0; c < 8; c++)
        reinterpret_cast<bf16x8*>(dst)[c] = *reinterpret_cast<bf16x8*>(&obuf[c * 8]);
}

// ------------------------------------------------------------- flash attention
// grid: (S/128, H). block 256. Wave w owns q rows [qt*128+w*16, +16), full key range streamed.
__global__ __launch_bounds__(256) void attn_kernel(
    const __bf16* __restrict__ Qb, const __bf16* __restrict__ Kb,
    const __bf16* __restrict__ Vt, __bf16* __restrict__ Ob)
{
    __shared__ __bf16 Pst[8][16 * 136];   // per-wave staging (16 q rows x 128, pad 8)

    const int lane = threadIdx.x & 31;
    const int wave = threadIdx.x >> 5;
    const int qt = blockIdx.x, h = blockIdx.y;
    const int qrow = qt * 128 + wave * 16;

    const __bf16* Q = Qb + ((size_t)h * S_LEN + qrow) * HD_DIM;
    const __bf16* K = Kb + (size_t)h * S_LEN * HD_DIM;
    const __bf16* V = Vt + (size_t)h * HD_DIM * S_LEN;   // [d][s]
    __bf16* pw = &Pst[wave][0];

    const float scale = 0.08838834764831845f;   // 1/sqrt(128)
    const int nlane = lane & 15, hi = lane >> 4;

    bf16x16 aQ[4];
#pragma unroll
    for (int kd = 0; kd < 4; kd++) aQ[kd] = load_a_frag(Q, HD_DIM, kd * 32, lane);

    f32x8 O[8] = {};
    float mrow[8], lrow[8];
#pragma unroll
    for (int v = 0; v < 8; v++) { mrow[v] = -3.0e38f; lrow[v] = 0.0f; }

    for (int kt = 0; kt < S_LEN / 128; kt++) {
        // Prefetch next K/V tiles one step ahead (global_prefetch_b8).
        if (kt + 1 < S_LEN / 128) {
            const char* kp = (const char*)(K + (size_t)(kt + 1) * 128 * HD_DIM) +
                             (size_t)threadIdx.x * 128;
            __builtin_prefetch(kp, 0, 3);
            const __bf16* vp = V + (size_t)(threadIdx.x >> 1) * S_LEN + (kt + 1) * 128 +
                               (threadIdx.x & 1) * 64;
            __builtin_prefetch(vp, 0, 3);
        }

        // S = Q K^T for this 128-key tile (double-buffered B fragments).
        f32x8 sc[8] = {};
        const __bf16* Kt = K + (size_t)kt * 128 * HD_DIM;
        bf16x16 bK0[4], bK1[4];
#pragma unroll
        for (int kd = 0; kd < 4; kd++) bK0[kd] = load_b_frag(Kt, HD_DIM, kd * 32, lane);
#pragma unroll
        for (int nt = 0; nt < 8; nt += 2) {
            const __bf16* Kn1 = Kt + (size_t)(nt + 1) * 16 * HD_DIM;
#pragma unroll
            for (int kd = 0; kd < 4; kd++) bK1[kd] = load_b_frag(Kn1, HD_DIM, kd * 32, lane);
#pragma unroll
            for (int kd = 0; kd < 4; kd++) sc[nt] = wmma_bf16(aQ[kd], bK0[kd], sc[nt]);
            if (nt + 2 < 8) {
                const __bf16* Kn2 = Kt + (size_t)(nt + 2) * 16 * HD_DIM;
#pragma unroll
                for (int kd = 0; kd < 4; kd++) bK0[kd] = load_b_frag(Kn2, HD_DIM, kd * 32, lane);
            }
#pragma unroll
            for (int kd = 0; kd < 4; kd++) sc[nt + 1] = wmma_bf16(aQ[kd], bK1[kd], sc[nt + 1]);
        }

        // online softmax row stats (row M=v+8*hi lives in this lane half)
        float mnew[8], fac[8];
#pragma unroll
        for (int v = 0; v < 8; v++) {
            float t = sc[0][v];
#pragma unroll
            for (int nt = 1; nt < 8; nt++) t = fmaxf(t, sc[nt][v]);
            t = fmaxf(t, __shfl_xor(t, 1, 16));
            t = fmaxf(t, __shfl_xor(t, 2, 16));
            t = fmaxf(t, __shfl_xor(t, 4, 16));
            t = fmaxf(t, __shfl_xor(t, 8, 16));
            mnew[v] = fmaxf(mrow[v], t * scale);
            fac[v] = __expf(mrow[v] - mnew[v]);
            mrow[v] = mnew[v];
        }
        // P = exp(S*scale - m): stage to LDS (wave-private; LDS is in-order per wave)
#pragma unroll
        for (int v = 0; v < 8; v++) {
            float rs = 0.0f;
#pragma unroll
            for (int nt = 0; nt < 8; nt++) {
                float p = __expf(sc[nt][v] * scale - mnew[v]);
                rs += p;
                pw[(v + 8 * hi) * 136 + nt * 16 + nlane] = (__bf16)p;
            }
            rs += __shfl_xor(rs, 1, 16);
            rs += __shfl_xor(rs, 2, 16);
            rs += __shfl_xor(rs, 4, 16);
            rs += __shfl_xor(rs, 8, 16);
            lrow[v] = lrow[v] * fac[v] + rs;
        }
        // rescale O
#pragma unroll
        for (int nt = 0; nt < 8; nt++)
#pragma unroll
            for (int v = 0; v < 8; v++) O[nt][v] *= fac[v];

        // O += P V  (A from LDS stage; B from pre-transposed V, double-buffered)
        bf16x16 aP[4];
#pragma unroll
        for (int kd = 0; kd < 4; kd++) aP[kd] = load_a_frag(pw, 136, kd * 32, lane);
        const __bf16* Vk = V + (size_t)kt * 128;
        bf16x16 bV0[4], bV1[4];
#pragma unroll
        for (int kd = 0; kd < 4; kd++) bV0[kd] = load_b_frag(Vk, S_LEN, kd * 32, lane);
#pragma unroll
        for (int nt = 0; nt < 8; nt += 2) {
            const __bf16* Vn1 = Vk + (size_t)(nt + 1) * 16 * S_LEN;
#pragma unroll
            for (int kd = 0; kd < 4; kd++) bV1[kd] = load_b_frag(Vn1, S_LEN, kd * 32, lane);
#pragma unroll
            for (int kd = 0; kd < 4; kd++) O[nt] = wmma_bf16(aP[kd], bV0[kd], O[nt]);
            if (nt + 2 < 8) {
                const __bf16* Vn2 = Vk + (size_t)(nt + 2) * 16 * S_LEN;
#pragma unroll
                for (int kd = 0; kd < 4; kd++) bV0[kd] = load_b_frag(Vn2, S_LEN, kd * 32, lane);
            }
#pragma unroll
            for (int kd = 0; kd < 4; kd++) O[nt + 1] = wmma_bf16(aP[kd], bV1[kd], O[nt + 1]);
        }
    }

    // finalize: O /= l, route through the wave-private LDS stage for coalesced stores
#pragma unroll
    for (int v = 0; v < 8; v++) lrow[v] = 1.0f / lrow[v];
#pragma unroll
    for (int nt = 0; nt < 8; nt++)
#pragma unroll
        for (int v = 0; v < 8; v++)
            pw[(v + 8 * hi) * 136 + nt * 16 + nlane] = (__bf16)(O[nt][v] * lrow[v]);

    const int rr = lane & 15, half = lane >> 4;
    const __bf16* srcr = pw + rr * 136 + half * 64;
    __bf16* dsto = Ob + (size_t)(qrow + rr) * C_DIM + h * HD_DIM + half * 64;
#pragma unroll
    for (int c = 0; c < 8; c++)
        reinterpret_cast<bf16x8*>(dsto)[c] = *reinterpret_cast<const bf16x8*>(srcr + c * 8);
}

// ------------------------------------------------------------- output projection
// grid: (S/128, C/128). out = Ob @ wo^T + bo, fp32, coalesced through LDS.
__global__ __launch_bounds__(256) void proj_kernel(
    const __bf16* __restrict__ Ab, const __bf16* __restrict__ Wb,
    const float* __restrict__ bo, float* __restrict__ out)
{
    __shared__ float tile[128 * 128];

    const int lane = threadIdx.x & 31;
    const int wave = threadIdx.x >> 5;
    const int mt = blockIdx.x, ntb = blockIdx.y;

    const __bf16* Arow = Ab + (size_t)(mt * 128 + wave * 16) * C_DIM;
    const __bf16* Brow = Wb + (size_t)(ntb * 128) * C_DIM;

    f32x8 acc[8] = {};
    bf16x16 aF0, bF0[8];
    aF0 = load_a_frag(Arow, C_DIM, 0, lane);
#pragma unroll
    for (int nt = 0; nt < 8; nt++) bF0[nt] = load_b_frag(Brow + (size_t)nt * 16 * C_DIM, C_DIM, 0, lane);

    for (int k0 = 0; k0 < C_DIM; k0 += 64) {
        bf16x16 aF1 = load_a_frag(Arow, C_DIM, k0 + 32, lane);
        bf16x16 bF1[8];
#pragma unroll
        for (int nt = 0; nt < 8; nt++)
            bF1[nt] = load_b_frag(Brow + (size_t)nt * 16 * C_DIM, C_DIM, k0 + 32, lane);
#pragma unroll
        for (int nt = 0; nt < 8; nt++) acc[nt] = wmma_bf16(aF0, bF0[nt], acc[nt]);
        if (k0 + 64 < C_DIM) {
            aF0 = load_a_frag(Arow, C_DIM, k0 + 64, lane);
#pragma unroll
            for (int nt = 0; nt < 8; nt++)
                bF0[nt] = load_b_frag(Brow + (size_t)nt * 16 * C_DIM, C_DIM, k0 + 64, lane);
        }
#pragma unroll
        for (int nt = 0; nt < 8; nt++) acc[nt] = wmma_bf16(aF1, bF1[nt], acc[nt]);
    }

    const int nlane = lane & 15, hi = lane >> 4;
#pragma unroll
    for (int nt = 0; nt < 8; nt++)
#pragma unroll
        for (int v = 0; v < 8; v++)
            tile[(wave * 16 + v + 8 * hi) * 128 + nt * 16 + nlane] = acc[nt][v];
    __syncthreads();

    const int r = threadIdx.x >> 1, half = threadIdx.x & 1;
    const float* rowp = &tile[r * 128 + half * 64];
    const float* bp = bo + ntb * 128 + half * 64;
    float* dst = out + (size_t)(mt * 128 + r) * C_DIM + ntb * 128 + half * 64;
#pragma unroll
    for (int c = 0; c < 16; c++) {
        float4 o;
        o.x = rowp[c * 4 + 0] + bp[c * 4 + 0];
        o.y = rowp[c * 4 + 1] + bp[c * 4 + 1];
        o.z = rowp[c * 4 + 2] + bp[c * 4 + 2];
        o.w = rowp[c * 4 + 3] + bp[c * 4 + 3];
        reinterpret_cast<float4*>(dst)[c] = o;
    }
}

// ----------------------------------------------------------------------- launch
extern "C" void kernel_launch(void* const* d_in, const int* in_sizes, int n_in,
                              void* d_out, int out_size, void* d_ws, size_t ws_size,
                              hipStream_t stream) {
    const float* x    = (const float*)d_in[0];
    const float* rope = (const float*)d_in[1];
    const float* wq   = (const float*)d_in[2];
    const float* bq   = (const float*)d_in[3];
    const float* wk   = (const float*)d_in[4];
    const float* bk   = (const float*)d_in[5];
    const float* wv   = (const float*)d_in[6];
    const float* bv   = (const float*)d_in[7];
    const float* qw   = (const float*)d_in[8];
    const float* kw   = (const float*)d_in[9];
    const float* wo   = (const float*)d_in[10];
    const float* bo   = (const float*)d_in[11];

    const size_t E = (size_t)S_LEN * C_DIM;   // 4,194,304 elements
    __bf16* xb  = (__bf16*)d_ws;
    __bf16* wqb = xb  + E;
    __bf16* wkb = wqb + E;
    __bf16* wvb = wkb + E;
    __bf16* wob = wvb + E;
    __bf16* Qb  = wob + E;   // [h][s][d]
    __bf16* Kb  = Qb  + E;   // [h][s][d]
    __bf16* Vt  = Kb  + E;   // [h][d][s]
    __bf16* Ob  = Vt  + E;   // [s][c]

    const int n8 = (int)(E / 8);
    const int cb = n8 / 256;
    cvt_f32_bf16<<<cb, 256, 0, stream>>>(x,  xb,  n8);
    cvt_f32_bf16<<<cb, 256, 0, stream>>>(wq, wqb, n8);
    cvt_f32_bf16<<<cb, 256, 0, stream>>>(wk, wkb, n8);
    cvt_f32_bf16<<<cb, 256, 0, stream>>>(wv, wvb, n8);
    cvt_f32_bf16<<<cb, 256, 0, stream>>>(wo, wob, n8);

    qkv_kernel<<<dim3(S_LEN / 128, N_HEAD, 3), 256, 0, stream>>>(
        xb, wqb, wkb, wvb, bq, bk, bv, qw, kw, rope, Qb, Kb, Vt);

    attn_kernel<<<dim3(S_LEN / 128, N_HEAD), 256, 0, stream>>>(Qb, Kb, Vt, Ob);

    proj_kernel<<<dim3(S_LEN / 128, C_DIM / 128), 256, 0, stream>>>(Ob, wob, bo, (float*)d_out);
}
